// PCVAdapter_21268678050373
// MI455X (gfx1250) — matrix operand order repaired and checked by reading, not compiled, not run on hardware
//
#include <hip/hip_runtime.h>
#include <math.h>
#include <stdint.h>

// ---------------------------------------------------------------------------
// MI455X (gfx1250) implementation of the recurrent MoG-RAFT refinement net.
// All convolutions -> implicit-GEMM on v_wmma_f32_16x16x32_f16 (f16 in, f32
// accumulate), weights staged to LDS with GLOBAL_LOAD_ASYNC_TO_LDS_B128
// (ASYNCcnt + s_wait_asynccnt). Correlation volume -> batched WMMA GEMM.
// Pointwise / gather / resize ops on VALU (negligible FLOP share).
// Compute-bound workload (~475 GFLOP over ~100MB) => matrix pipe dominates.
// ---------------------------------------------------------------------------

typedef __attribute__((ext_vector_type(16))) _Float16 v16h;
typedef __attribute__((ext_vector_type(8)))  float    v8f;

// b128 async-copy payload type: matches builtin proto (int __vector(4) ptrs)
typedef int v4i_b128 __attribute__((vector_size(16)));
typedef __attribute__((address_space(1))) v4i_b128 as1_v4i;   // global
typedef __attribute__((address_space(3))) v4i_b128 as3_v4i;   // LDS

#define BH   2
#define HD4  120
#define WD4  160
#define HW4  (HD4*WD4)     // 19200
#define HD8  60
#define WD8  80
#define HW8  (HD8*WD8)     // 4800
#define HD16 30
#define WD16 40
#define HW16 (HD16*WD16)   // 1200

static __host__ __device__ inline int ceil_div_i(int a, int b) { return (a + b - 1) / b; }

// ---- gfx1250 async global->LDS copy (16B per lane) + ASYNCcnt wait --------
__device__ __forceinline__ void async_copy_b128(const float* gsrc, float* lds_dst)
{
#if __has_builtin(__builtin_amdgcn_global_load_async_to_lds_b128)
    __builtin_amdgcn_global_load_async_to_lds_b128(
        (as1_v4i*)(uintptr_t)gsrc,
        (as3_v4i*)(uint32_t)(uintptr_t)lds_dst,
        0, 0);
#else
    unsigned loff = (unsigned)(uintptr_t)lds_dst;     // low 32b of flat ptr = LDS offset
    unsigned long long gaddr = (unsigned long long)(uintptr_t)gsrc;
    asm volatile("global_load_async_to_lds_b128 %0, %1, off"
                 :: "v"(loff), "v"(gaddr) : "memory");
#endif
}

__device__ __forceinline__ void wait_async0()
{
#if __has_builtin(__builtin_amdgcn_s_wait_asynccnt)
    __builtin_amdgcn_s_wait_asynccnt(0);
#else
    asm volatile("s_wait_asynccnt 0x0" ::: "memory");
#endif
}

// ---------------------------------------------------------------------------
// Generic implicit-GEMM conv (1x1 or 3x3 SAME), NCHW fp32 in/out, f16 WMMA.
// Block: 128 threads = 4 waves. Block tile: 16 (Cout) x 128 (pixels);
// each wave owns 2 accumulators (16x32 pixels) and reuses one A fragment.
// K = Cin * (k3 ? 9 : 1), ordered k = ci*9 + tap (weight rows contiguous).
// A tile (16x32 f32, 2KB) staged via async DMA to LDS when fully in-bounds.
// mode: 0=linear 1=relu 2=sigmoid 3=tanh
// ---------------------------------------------------------------------------
__global__ __launch_bounds__(128)
void conv_wmma_k(const float* __restrict__ act, const float* __restrict__ wgt,
                 float* __restrict__ out,
                 int Cin, int Cout, int H, int W, int k3, int mode,
                 int out_off, int out_tot)
{
    const int HWp  = H * W;
    const int b    = blockIdx.z;
    const int coT  = blockIdx.y;
    const int lane = threadIdx.x & 31;
    const int wave = threadIdx.x >> 5;
    const int pBase = blockIdx.x * 128 + wave * 32;
    const int Ktot = Cin * (k3 ? 9 : 1);

    __shared__ __align__(16) float lds_a[16 * 32];

    const float* actB = act + (size_t)b * Cin * HWp;

    const int nLoc = lane & 15;
    const int half = lane >> 4;
    const int p0 = pBase + nLoc;
    const int p1 = p0 + 16;
    const int pc0 = (p0 < HWp) ? p0 : (HWp - 1);   // clamp, mask at store
    const int pc1 = (p1 < HWp) ? p1 : (HWp - 1);
    const int py0 = pc0 / W, px0 = pc0 - py0 * W;
    const int py1 = pc1 / W, px1 = pc1 - py1 * W;

    const bool tileMfull = (coT * 16 + 16) <= Cout;   // block-uniform

    v8f acc0 = {}, acc1 = {};

    for (int ks = 0; ks < Ktot; ks += 32) {
        __syncthreads();
        if (tileMfull && (ks + 32) <= Ktot) {
            // async DMA: 128 lanes x b128 = full 16x32 f32 weight tile -> LDS
            int m  = threadIdx.x >> 3;
            int ch = (threadIdx.x & 7) << 2;
            async_copy_b128(&wgt[(size_t)(coT * 16 + m) * Ktot + ks + ch],
                            &lds_a[(m << 5) + ch]);
            wait_async0();
        } else {
            int idx = threadIdx.x << 2;
#pragma unroll
            for (int t = 0; t < 4; ++t) {
                int ii = idx + t;
                int m  = ii >> 5, kk = ii & 31;
                int mg = coT * 16 + m;
                int k  = ks + kk;
                lds_a[(m << 5) + kk] =
                    (mg < Cout && k < Ktot) ? wgt[(size_t)mg * Ktot + k] : 0.0f;
            }
        }
        __syncthreads();
        if (ks + 32 < Ktot)
            __builtin_prefetch(&wgt[(size_t)(coT * 16) * Ktot + ks + 32], 0, 1);

        // A fragment: 16x32 f16, lanes 0-15 M=0..15 K 0-7/16-23, lanes 16-31 K 8-15/24-31
        v16h af;
#pragma unroll
        for (int e = 0; e < 16; ++e) {
            int v    = e >> 1, bit = e & 1;
            int koff = ((v < 4) ? (2 * v) : (16 + 2 * (v - 4))) + 8 * half + bit;
            af[e] = (_Float16)lds_a[(nLoc << 5) + koff];
        }

        // Two B fragments (32x16 each), on-the-fly im2col with zero padding.
        v16h bf0, bf1;
#pragma unroll
        for (int e = 0; e < 16; ++e) {
            int   k  = ks + ((e < 8) ? e : (8 + e)) + 8 * half;
            float v0 = 0.0f, v1 = 0.0f;
            if (k < Ktot) {
                if (k3) {
                    int ci  = k / 9;
                    int tap = k - ci * 9;
                    int dy  = tap / 3;
                    int dxx = tap - dy * 3;
                    int yy0 = py0 + dy - 1, xx0 = px0 + dxx - 1;
                    int yy1 = py1 + dy - 1, xx1 = px1 + dxx - 1;
                    if ((unsigned)yy0 < (unsigned)H && (unsigned)xx0 < (unsigned)W)
                        v0 = actB[(size_t)ci * HWp + yy0 * W + xx0];
                    if ((unsigned)yy1 < (unsigned)H && (unsigned)xx1 < (unsigned)W)
                        v1 = actB[(size_t)ci * HWp + yy1 * W + xx1];
                } else {
                    v0 = actB[(size_t)k * HWp + pc0];
                    v1 = actB[(size_t)k * HWp + pc1];
                }
            }
            bf0[e] = (_Float16)v0;
            bf1[e] = (_Float16)v1;
        }
        acc0 = __builtin_amdgcn_wmma_f32_16x16x32_f16(false, af, false, bf0,
                                                      (short)0, acc0, false, false);
        acc1 = __builtin_amdgcn_wmma_f32_16x16x32_f16(false, af, false, bf1,
                                                      (short)0, acc1, false, false);
    }

#pragma unroll
    for (int r = 0; r < 8; ++r) {
        int mg = coT * 16 + r + 8 * half;
        if (mg < Cout) {
            size_t rowOff = ((size_t)b * out_tot + out_off + mg) * HWp;
            float v = acc0[r];
            float u = acc1[r];
            if (mode == 1)      { v = fmaxf(v, 0.0f);               u = fmaxf(u, 0.0f); }
            else if (mode == 2) { v = 1.0f / (1.0f + __expf(-v));   u = 1.0f / (1.0f + __expf(-u)); }
            else if (mode == 3) { v = tanhf(v);                     u = tanhf(u); }
            if (p0 < HWp) out[rowOff + p0] = v;
            if (p1 < HWp) out[rowOff + p1] = u;
        }
    }
}

// ---------------------------------------------------------------------------
// Correlation: corr[b,h,i,j] = scale * sum_c f1[b,c,h,i]*f2[b,c,h,j]
// Batched GEMM per (b,h): [160 x 128] * [128 x 160].
// One wave per 16(i) x 32(j) tile: 2 accumulators, shared A fragment.
// ---------------------------------------------------------------------------
__global__ __launch_bounds__(32)
void corr_wmma_k(const float* __restrict__ f1, const float* __restrict__ f2,
                 float* __restrict__ corr, int C, int H, int W, float scale)
{
    const int bh    = blockIdx.z;      // b*H + h
    const int iTile = blockIdx.y;
    const int jBase = blockIdx.x * 32;
    const int lane  = threadIdx.x & 31;
    const int low   = lane & 15;
    const int half  = lane >> 4;

    const int b = bh / H, h = bh - b * H;
    const size_t base = ((size_t)b * C) * (size_t)(H * W) + (size_t)h * W;

    v8f acc0 = {}, acc1 = {};
    for (int ks = 0; ks < C; ks += 32) {
        v16h af, bf0, bf1;
#pragma unroll
        for (int e = 0; e < 16; ++e) {          // A[i,c] = f1[b,c,h,i]
            int v    = e >> 1, bit = e & 1;
            int koff = ((v < 4) ? (2 * v) : (16 + 2 * (v - 4))) + 8 * half + bit;
            int c    = ks + koff;
            af[e] = (_Float16)f1[base + (size_t)c * (H * W) + iTile * 16 + low];
        }
#pragma unroll
        for (int e = 0; e < 16; ++e) {          // B[c,j] = f2[b,c,h,j]
            int c = ks + ((e < 8) ? e : (8 + e)) + 8 * half;
            const float* rowp = &f2[base + (size_t)c * (H * W) + jBase + low];
            bf0[e] = (_Float16)rowp[0];
            bf1[e] = (_Float16)rowp[16];
        }
        acc0 = __builtin_amdgcn_wmma_f32_16x16x32_f16(false, af, false, bf0,
                                                      (short)0, acc0, false, false);
        acc1 = __builtin_amdgcn_wmma_f32_16x16x32_f16(false, af, false, bf1,
                                                      (short)0, acc1, false, false);
    }
#pragma unroll
    for (int r = 0; r < 8; ++r) {
        int ii = iTile * 16 + r + 8 * half;
        size_t rowOff = ((size_t)bh * W + ii) * W;
        corr[rowOff + jBase + low]      = acc0[r] * scale;
        corr[rowOff + jBase + 16 + low] = acc1[r] * scale;
    }
}

// ------------------------------- VALU kernels ------------------------------

__global__ void fill_k(float* __restrict__ p, float v, int n) {
    int i = blockIdx.x * blockDim.x + threadIdx.x;
    if (i < n) p[i] = v;
}

__global__ void init_state_k(float* __restrict__ mu, float* __restrict__ sg,
                             float* __restrict__ wv, int n) {
    int i = blockIdx.x * blockDim.x + threadIdx.x;
    if (i < n) { mu[i] = 0.0f; sg[i] = 0.25f; wv[i] = 1.0f / 3.0f; }
}

// pool along last (j) dim of [rows, 2*Wout] -> [rows, Wout]
__global__ void pool_last_k(const float* __restrict__ in, float* __restrict__ out,
                            int rows, int Wout) {
    int i = blockIdx.x * blockDim.x + threadIdx.x;
    int total = rows * Wout;
    if (i >= total) return;
    int r = i / Wout, j = i - r * Wout;
    const float* ip = in + (size_t)r * (Wout * 2) + 2 * j;
    out[i] = 0.5f * (ip[0] + ip[1]);
}

// MoG sampling: corr_feat[b, l*27+k*9+s, h, w]
__global__ void sample_k(const float* __restrict__ p0, const float* __restrict__ p1,
                         const float* __restrict__ p2, const float* __restrict__ p3,
                         const float* __restrict__ mu, const float* __restrict__ sg,
                         float* __restrict__ outF) {
    int i = blockIdx.x * blockDim.x + threadIdx.x;
    const int total = BH * 3 * 9 * HW4;
    if (i >= total) return;
    int w = i % WD4; int t = i / WD4;
    int h = t % HD4; t /= HD4;
    int s = t % 9;   t /= 9;
    int k = t % 3;   int b = t / 3;
    size_t sp  = ((size_t)(b * 3 + k)) * HW4 + (size_t)h * WD4 + w;
    float muv  = mu[sp], sgv = sg[sp];
    float pos  = ((float)w - muv) + sgv * (float)(s - 4);
    const float* pyr[4] = { p0, p1, p2, p3 };
    size_t rowBase = (size_t)(b * HD4 + h) * WD4 + w;
#pragma unroll
    for (int l = 0; l < 4; ++l) {
        int Wl = WD4 >> l;
        const float* row = pyr[l] + rowBase * Wl;
        float x  = pos / (float)(1 << l);
        float x0 = fminf(fmaxf(floorf(x), 0.0f), (float)(Wl - 2));
        float f  = fminf(fmaxf(x - x0, 0.0f), 1.0f);
        int  i0  = (int)x0;
        float v0 = row[i0], v1 = row[i0 + 1];
        int c = l * 27 + k * 9 + s;
        outF[((size_t)b * 108 + c) * HW4 + (size_t)h * WD4 + w] = v0 + f * (v1 - v0);
    }
}

// 2x2 average pool, writes into channel slice [off, off+C) of dst (tot ch)
__global__ void pool2_k(const float* __restrict__ in, float* __restrict__ out,
                        int C, int Hi, int Wi, int off, int tot) {
    int Ho = Hi >> 1, Wo = Wi >> 1;
    int i = blockIdx.x * blockDim.x + threadIdx.x;
    int total = BH * C * Ho * Wo;
    if (i >= total) return;
    int x = i % Wo; int t = i / Wo;
    int y = t % Ho; t /= Ho;
    int c = t % C;  int b = t / C;
    const float* ip = in + (((size_t)(b * C + c) * Hi + 2 * y) * Wi + 2 * x);
    float v = 0.25f * (ip[0] + ip[1] + ip[Wi] + ip[Wi + 1]);
    out[((size_t)b * tot + off + c) * (Ho * Wo) + (size_t)y * Wo + x] = v;
}

// 2x bilinear upsample (half-pixel centers, edge clamp)
__global__ void up2_k(const float* __restrict__ in, float* __restrict__ out,
                      int C, int Hi, int Wi, int off, int tot) {
    int Ho = Hi * 2, Wo = Wi * 2;
    int i = blockIdx.x * blockDim.x + threadIdx.x;
    int total = BH * C * Ho * Wo;
    if (i >= total) return;
    int x = i % Wo; int t = i / Wo;
    int y = t % Ho; t /= Ho;
    int c = t % C;  int b = t / C;
    float syf = y * 0.5f - 0.25f;
    float sxf = x * 0.5f - 0.25f;
    float y0f = floorf(syf), x0f = floorf(sxf);
    float fy = syf - y0f,   fx = sxf - x0f;
    int y0 = (int)y0f, x0 = (int)x0f;
    int y0a = min(max(y0, 0), Hi - 1), y1a = min(max(y0 + 1, 0), Hi - 1);
    int x0a = min(max(x0, 0), Wi - 1), x1a = min(max(x0 + 1, 0), Wi - 1);
    const float* bp = in + (size_t)(b * C + c) * (Hi * Wi);
    float v00 = bp[(size_t)y0a * Wi + x0a], v01 = bp[(size_t)y0a * Wi + x1a];
    float v10 = bp[(size_t)y1a * Wi + x0a], v11 = bp[(size_t)y1a * Wi + x1a];
    float v0 = v00 + fx * (v01 - v00);
    float v1 = v10 + fx * (v11 - v10);
    out[((size_t)b * tot + off + c) * (Ho * Wo) + (size_t)y * Wo + x] = v0 + fy * (v1 - v0);
}

// copy C channels src[srcOff..] -> dst[dstOff..]
__global__ void copy_ch_k(const float* __restrict__ src, float* __restrict__ dst,
                          int C, int HWp, int srcTot, int srcOff, int dstTot, int dstOff) {
    int i = blockIdx.x * blockDim.x + threadIdx.x;
    int total = BH * C * HWp;
    if (i >= total) return;
    int p = i % HWp; int t = i / HWp;
    int c = t % C;   int b = t / C;
    dst[((size_t)b * dstTot + dstOff + c) * HWp + p] =
        src[((size_t)b * srcTot + srcOff + c) * HWp + p];
}

// hx[ch 0..127] = r * net (elementwise)
__global__ void rmulh_k(const float* __restrict__ r, const float* __restrict__ net,
                        float* __restrict__ hx, int HWp, int hxTot) {
    int i = blockIdx.x * blockDim.x + threadIdx.x;
    int total = BH * 128 * HWp;
    if (i >= total) return;
    int p = i % HWp; int t = i / HWp;
    int c = t % 128; int b = t / 128;
    size_t si = ((size_t)b * 128 + c) * HWp + p;
    hx[((size_t)b * hxTot + c) * HWp + p] = r[si] * net[si];
}

// net = (1-z)*net + z*q
__global__ void gru_upd_k(float* __restrict__ net, const float* __restrict__ z,
                          const float* __restrict__ q, int n) {
    int i = blockIdx.x * blockDim.x + threadIdx.x;
    if (i < n) { float zv = z[i]; net[i] = (1.0f - zv) * net[i] + zv * q[i]; }
}

// out[b, 0..2]=a, 3..5=b2, 6..8=c2 ; each src is [BH,3,HW4]
__global__ void pack3_k(const float* __restrict__ a, const float* __restrict__ b2,
                        const float* __restrict__ c2, float* __restrict__ out) {
    int i = blockIdx.x * blockDim.x + threadIdx.x;
    int total = BH * 9 * HW4;
    if (i >= total) return;
    int p = i % HW4; int t = i / HW4;
    int ch = t % 9;  int bb = t / 9;
    const float* src = (ch < 3) ? a : ((ch < 6) ? b2 : c2);
    int k = ch % 3;
    out[i] = src[((size_t)bb * 3 + k) * HW4 + p];
}

// mu += dmu ; sigma = softplus(sigma + dsig) ; w = softmax(wlog)
__global__ void update_k(float* __restrict__ mu, float* __restrict__ sg,
                         float* __restrict__ wv, const float* __restrict__ delta) {
    int i = blockIdx.x * blockDim.x + threadIdx.x;
    const int total = BH * HW4;
    if (i >= total) return;
    int p = i % HW4; int b = i / HW4;
    const float* d = delta + (size_t)b * 9 * HW4;
    float wl[3]; float wmax = -1e30f;
#pragma unroll
    for (int k = 0; k < 3; ++k) {
        size_t si = ((size_t)b * 3 + k) * HW4 + p;
        mu[si] += d[(size_t)k * HW4 + p];
        float s = sg[si] + d[(size_t)(3 + k) * HW4 + p];
        sg[si] = (s > 30.0f) ? s : log1pf(__expf(s));
        wl[k] = d[(size_t)(6 + k) * HW4 + p];
        wmax = fmaxf(wmax, wl[k]);
    }
    float e0 = __expf(wl[0] - wmax), e1 = __expf(wl[1] - wmax), e2 = __expf(wl[2] - wmax);
    float inv = 1.0f / (e0 + e1 + e2);
    size_t si = ((size_t)b * 3) * HW4 + p;
    wv[si] = e0 * inv; wv[si + HW4] = e1 * inv; wv[si + 2 * (size_t)HW4] = e2 * inv;
}

// per-channel mean / rstd over (B,H,W); y is [BH,32,HW4]
__global__ void bn_stats_k(const float* __restrict__ y, float* __restrict__ stats) {
    __shared__ float ssum[256], ssq[256];
    int c = blockIdx.x, tid = threadIdx.x;
    float s = 0.0f, q = 0.0f;
    for (int i = tid; i < BH * HW4; i += 256) {
        int b = i / HW4, p = i - b * HW4;
        float v = y[((size_t)b * 32 + c) * HW4 + p];
        s += v; q += v * v;
    }
    ssum[tid] = s; ssq[tid] = q; __syncthreads();
    for (int st = 128; st > 0; st >>= 1) {
        if (tid < st) { ssum[tid] += ssum[tid + st]; ssq[tid] += ssq[tid + st]; }
        __syncthreads();
    }
    if (tid == 0) {
        float n = (float)(BH * HW4);
        float mean = ssum[0] / n;
        float var  = ssq[0] / n - mean * mean;
        stats[c]      = mean;
        stats[32 + c] = rsqrtf(var + 1e-5f);
    }
}

__global__ void bn_apply_k(const float* __restrict__ y, const float* __restrict__ stats,
                           const float* __restrict__ gamma, const float* __restrict__ beta,
                           float* __restrict__ out) {
    int i = blockIdx.x * blockDim.x + threadIdx.x;
    int total = BH * 32 * HW4;
    if (i >= total) return;
    int t = i / HW4; int c = t % 32;
    float v = (y[i] - stats[c]) * stats[32 + c] * gamma[c] + beta[c];
    out[i] = fmaxf(v, 0.0f);
}

// ---------------------------------------------------------------------------
// Host orchestration
// ---------------------------------------------------------------------------
extern "C" void kernel_launch(void* const* d_in, const int* in_sizes, int n_in,
                              void* d_out, int out_size, void* d_ws, size_t ws_size,
                              hipStream_t stream)
{
    (void)n_in; (void)out_size; (void)ws_size;

    // ---- input pointer mapping (handles insertion order vs sorted pytree) ----
    const float *f4L, *f4R, *mf08, *W_f4, *W_ec, *W_ef, *W_eo;
    const float *g16z, *g16r, *g16q, *g08z, *g08r, *g08q, *g04z, *g04r, *g04q;
    const float *Wh1, *Wh2, *Wproj, *bnG, *bnB;
    if (in_sizes[2] == 1) {   // jax sorted-key flattening: iters at slot 2
        f4L  = (const float*)d_in[0];  f4R  = (const float*)d_in[1];
        mf08 = (const float*)d_in[3];
        bnB  = (const float*)d_in[4];  bnG  = (const float*)d_in[5];
        W_ec = (const float*)d_in[6];  W_ef = (const float*)d_in[7];  W_eo = (const float*)d_in[8];
        W_f4 = (const float*)d_in[9];
        g04q = (const float*)d_in[10]; g04r = (const float*)d_in[11]; g04z = (const float*)d_in[12];
        g08q = (const float*)d_in[13]; g08r = (const float*)d_in[14]; g08z = (const float*)d_in[15];
        g16q = (const float*)d_in[16]; g16r = (const float*)d_in[17]; g16z = (const float*)d_in[18];
        Wh1  = (const float*)d_in[19]; Wh2  = (const float*)d_in[20]; Wproj = (const float*)d_in[21];
    } else {                  // dict insertion order
        f4L  = (const float*)d_in[0];  f4R  = (const float*)d_in[1];
        mf08 = (const float*)d_in[2];  W_f4 = (const float*)d_in[3];
        W_ec = (const float*)d_in[4];  W_ef = (const float*)d_in[5];  W_eo = (const float*)d_in[6];
        g16z = (const float*)d_in[7];  g16r = (const float*)d_in[8];  g16q = (const float*)d_in[9];
        g08z = (const float*)d_in[10]; g08r = (const float*)d_in[11]; g08q = (const float*)d_in[12];
        g04z = (const float*)d_in[13]; g04r = (const float*)d_in[14]; g04q = (const float*)d_in[15];
        Wh1  = (const float*)d_in[16]; Wh2  = (const float*)d_in[17]; Wproj = (const float*)d_in[18];
        bnG  = (const float*)d_in[19]; bnB  = (const float*)d_in[20];
    }

    // ---- workspace layout ----
    float* wsf = (float*)d_ws;
    size_t off = 0;
    auto alloc = [&](size_t n) { float* q = wsf + off; off += n; return q; };
    float* fmap1 = alloc((size_t)BH * 128 * HW4);
    float* fmap2 = alloc((size_t)BH * 128 * HW4);
    float* pyr0  = alloc((size_t)BH * HD4 * WD4 * 160);
    float* pyr1  = alloc((size_t)BH * HD4 * WD4 * 80);
    float* pyr2  = alloc((size_t)BH * HD4 * WD4 * 40);
    float* pyr3  = alloc((size_t)BH * HD4 * WD4 * 20);
    float* cfeat = alloc((size_t)BH * 108 * HW4);
    float* cfff  = alloc((size_t)BH * 128 * HW4);   // cf(0-95)+ff(96-127); later head1 out
    float* flow9 = alloc((size_t)BH * 9 * HW4);     // also final feat
    float* mu    = alloc((size_t)BH * 3 * HW4);
    float* sigma = alloc((size_t)BH * 3 * HW4);
    float* wv    = alloc((size_t)BH * 3 * HW4);
    float* net04 = alloc((size_t)BH * 128 * HW4);
    float* net08 = alloc((size_t)BH * 128 * HW8);
    float* net16 = alloc((size_t)BH * 128 * HW16);
    float* hx04  = alloc((size_t)BH * 384 * HW4);
    float* hx08  = alloc((size_t)BH * 512 * HW8);
    float* hx16  = alloc((size_t)BH * 256 * HW16);
    float* zbuf  = alloc((size_t)BH * 128 * HW4);   // shared across scales
    float* rbuf  = alloc((size_t)BH * 128 * HW4);   // shared across scales (also q)
    float* delta = alloc((size_t)BH * 9 * HW4);
    float* ybuf  = alloc((size_t)BH * 32 * HW4);
    float* stats = alloc(64);

    auto l1 = [&](int n) { return dim3((unsigned)ceil_div_i(n, 256)); };
    auto conv = [&](const float* act, const float* w, float* out_, int Cin, int Cout,
                    int H, int W, int k3, int mode, int oOff, int oTot) {
        dim3 grid((unsigned)ceil_div_i(H * W, 128), (unsigned)ceil_div_i(Cout, 16), BH);
        conv_wmma_k<<<grid, 128, 0, stream>>>(act, w, out_, Cin, Cout, H, W, k3, mode, oOff, oTot);
    };
    auto gru = [&](float* net, float* hx, int hxC, int H, int W,
                   const float* wz, const float* wr, const float* wq) {
        int HWp = H * W, nEl = BH * 128 * HWp;
        copy_ch_k<<<l1(nEl), 256, 0, stream>>>(net, hx, 128, HWp, 128, 0, hxC, 0);
        conv(hx, wz, zbuf, hxC, 128, H, W, 1, 2, 0, 128);          // z = sigmoid
        conv(hx, wr, rbuf, hxC, 128, H, W, 1, 2, 0, 128);          // r = sigmoid
        rmulh_k<<<l1(nEl), 256, 0, stream>>>(rbuf, net, hx, HWp, hxC);  // hx[0:128] = r*h
        conv(hx, wq, rbuf, hxC, 128, H, W, 1, 3, 0, 128);          // q = tanh
        gru_upd_k<<<l1(nEl), 256, 0, stream>>>(net, zbuf, rbuf, nEl);
    };

    // ---- feature projection + correlation pyramid (once) ----
    conv(f4L, W_f4, fmap1, 512, 128, HD4, WD4, 0, 0, 0, 128);
    conv(f4R, W_f4, fmap2, 512, 128, HD4, WD4, 0, 0, 0, 128);
    {
        dim3 g(5, 10, BH * HD4);
        corr_wmma_k<<<g, 32, 0, stream>>>(fmap1, fmap2, pyr0, 128, HD4, WD4,
                                          0.08838834764831845f); // 1/sqrt(128)
    }
    {
        int rows = BH * HD4 * WD4;
        pool_last_k<<<l1(rows * 80), 256, 0, stream>>>(pyr0, pyr1, rows, 80);
        pool_last_k<<<l1(rows * 40), 256, 0, stream>>>(pyr1, pyr2, rows, 40);
        pool_last_k<<<l1(rows * 20), 256, 0, stream>>>(pyr2, pyr3, rows, 20);
    }

    // ---- state init ----
    fill_k<<<l1(BH * 128 * HW4), 256, 0, stream>>>(net04, 0.0f, BH * 128 * HW4);
    fill_k<<<l1(BH * 128 * HW8), 256, 0, stream>>>(net08, 0.0f, BH * 128 * HW8);
    fill_k<<<l1(BH * 128 * HW16), 256, 0, stream>>>(net16, 0.0f, BH * 128 * HW16);
    init_state_k<<<l1(BH * 3 * HW4), 256, 0, stream>>>(mu, sigma, wv, BH * 3 * HW4);
    // constant mf_08 -> hx08 channels [384,512)
    copy_ch_k<<<l1(BH * 128 * HW8), 256, 0, stream>>>(mf08, hx08, 128, HW8, 128, 0, 512, 384);

    // ---- 3 refinement iterations (fixed count for graph capture) ----
    for (int it = 0; it < 3; ++it) {
        // sample correlation features
        sample_k<<<l1(BH * 27 * HW4), 256, 0, stream>>>(pyr0, pyr1, pyr2, pyr3,
                                                        mu, sigma, cfeat);
        // encoders
        conv(cfeat, W_ec, cfff, 108, 96, HD4, WD4, 0, 1, 0, 128);          // cf
        pack3_k<<<l1(BH * 9 * HW4), 256, 0, stream>>>(mu, sigma, wv, flow9);
        conv(flow9, W_ef, cfff, 9, 32, HD4, WD4, 0, 1, 96, 128);           // ff
        conv(cfff, W_eo, hx04, 128, 128, HD4, WD4, 0, 1, 128, 384);        // mf04 -> hx04[128:256]

        // GRU @1/16
        pool2_k<<<l1(BH * 128 * HW16), 256, 0, stream>>>(net08, hx16, 128, HD8, WD8, 128, 256);
        gru(net16, hx16, 256, HD16, WD16, g16z, g16r, g16q);
        // GRU @1/8
        pool2_k<<<l1(BH * 128 * HW8), 256, 0, stream>>>(net04, hx08, 128, HD4, WD4, 128, 512);
        up2_k<<<l1(BH * 128 * HW8), 256, 0, stream>>>(net16, hx08, 128, HD16, WD16, 256, 512);
        gru(net08, hx08, 512, HD8, WD8, g08z, g08r, g08q);
        // GRU @1/4
        up2_k<<<l1(BH * 128 * HW4), 256, 0, stream>>>(net08, hx04, 128, HD8, WD8, 256, 384);
        gru(net04, hx04, 384, HD4, WD4, g04z, g04r, g04q);

        // head + MoG update
        conv(net04, Wh1, cfff, 128, 128, HD4, WD4, 1, 1, 0, 128);
        conv(cfff, Wh2, delta, 128, 9, HD4, WD4, 1, 0, 0, 9);
        update_k<<<l1(BH * HW4), 256, 0, stream>>>(mu, sigma, wv, delta);
    }

    // ---- projection + batchnorm + relu ----
    pack3_k<<<l1(BH * 9 * HW4), 256, 0, stream>>>(wv, mu, sigma, flow9); // [w,mu,sigma]
    conv(flow9, Wproj, ybuf, 9, 32, HD4, WD4, 0, 0, 0, 32);
    bn_stats_k<<<32, 256, 0, stream>>>(ybuf, stats);
    bn_apply_k<<<l1(BH * 32 * HW4), 256, 0, stream>>>(ybuf, stats, bnG, bnB, (float*)d_out);
}
